// SimpleRNN_18021682774625
// MI455X (gfx1250) — matrix operand order repaired and checked by reading, not compiled, not run on hardware
//
#include <hip/hip_runtime.h>

// ---------------- types for WMMA ----------------
typedef __attribute__((ext_vector_type(16))) __bf16 v16bf;
typedef __attribute__((ext_vector_type(8)))  __bf16 v8bf;
typedef __attribute__((ext_vector_type(8)))  float  v8f;

// ---------------- problem constants ----------------
#define TT   2048      // sequence length
#define EE   300       // embedding dim
#define EP   320       // padded K (multiple of 32)
#define HH   512       // hidden
#define H4   2048      // 4*H

// workspace layout (bytes, 256-aligned)
#define OFF_X    ((size_t)0)                         // 2*T*EP bf16      = 2,621,440
#define OFF_WIH  ((size_t)2621440)                   // 2*H4*EP bf16     = 2,621,440
#define OFF_WHH  ((size_t)5242880)                   // 2*H4*H bf16      = 4,194,304
#define OFF_GX   ((size_t)9437184)                   // 4*T*H4 f32       = 67,108,864
#define OFF_HOUT ((size_t)76546048)                  // 4*H f32

// ---------------- helpers ----------------
__device__ __forceinline__ unsigned short f2bf(float x) {
    unsigned u = __float_as_uint(x);
    unsigned r = u + 0x7FFFu + ((u >> 16) & 1u);     // RNE
    return (unsigned short)(r >> 16);
}
__device__ __forceinline__ float sigf(float x) { return 1.0f / (1.0f + expf(-x)); }

// ---------------- prep: w_ih f32[2,4H,300] -> bf16[2,4H,320] (zero pad) ----------------
__global__ void k_conv_wih(const float* __restrict__ w_ih, unsigned short* __restrict__ out) {
    int idx = blockIdx.x * blockDim.x + threadIdx.x;
    if (idx >= 2 * H4 * EP) return;
    int k = idx % EP;
    int r = (idx / EP) % H4;
    int d = idx / (EP * H4);
    float v = (k < EE) ? w_ih[((size_t)d * H4 + r) * EE + k] : 0.0f;
    out[idx] = f2bf(v);
}

// ---------------- prep: w_hh f32[2,4H,512] -> bf16 ----------------
__global__ void k_conv_whh(const float* __restrict__ w_hh, unsigned short* __restrict__ out) {
    int idx = blockIdx.x * blockDim.x + threadIdx.x;
    if (idx >= 2 * H4 * HH) return;
    out[idx] = f2bf(w_hh[idx]);
}

// ---------------- prep: gather embeddings -> bf16 X[2, T, EP] ----------------
__global__ void k_embed(const int* __restrict__ sentA, const int* __restrict__ sentB,
                        const float* __restrict__ emb, unsigned short* __restrict__ X) {
    int idx = blockIdx.x * blockDim.x + threadIdx.x;
    if (idx >= 2 * TT * EP) return;
    int k = idx % EP;
    int t = (idx / EP) % TT;
    int s = idx / (EP * TT);
    int tok = (s == 0 ? sentA : sentB)[t];
    float v = (k < EE) ? emb[(size_t)tok * EE + k] : 0.0f;
    X[idx] = f2bf(v);
}

// ---------------- WMMA GEMM: gx[combo, T, 4H] = X[s] @ Wih[d]^T ----------------
__global__ __launch_bounds__(128) void k_gemm_gx(const unsigned short* __restrict__ X,
                                                 const unsigned short* __restrict__ Wih,
                                                 float* __restrict__ gx) {
    const int lane = threadIdx.x & 31;
    const int wave = threadIdx.x >> 5;
    int tile  = blockIdx.x * 4 + wave;           // 0 .. 65535
    int combo = tile >> 14;                       // 4 combos x 128x128 tiles
    int rem   = tile & 16383;
    int mTile = rem >> 7;
    int nTile = rem & 127;
    int s = combo >> 1, d = combo & 1;

    const unsigned short* Xs = X   + (size_t)s * TT * EP;
    const unsigned short* Wd = Wih + (size_t)d * H4 * EP;

    const int l15  = lane & 15;
    const int half = lane >> 4;
    int mrow = (mTile << 4) + l15;
    int ncol = (nTile << 4) + l15;
    const unsigned short* arow = Xs + (size_t)mrow * EP + half * 16;
    const unsigned short* brow = Wd + (size_t)ncol * EP + half * 16;

    v8f acc = {};
#pragma unroll
    for (int kb = 0; kb < EP; kb += 32) {
        v8bf alo = *(const v8bf*)(arow + kb);
        v8bf ahi = *(const v8bf*)(arow + kb + 8);
        v16bf a = __builtin_shufflevector(alo, ahi,
                    0,1,2,3,4,5,6,7,8,9,10,11,12,13,14,15);
        v16bf b = *(const v16bf*)(brow + kb);
        acc = __builtin_amdgcn_wmma_f32_16x16x32_bf16(
                  false, a, false, b, (short)0, acc, false, false);
    }

    float* out = gx + (size_t)combo * TT * H4
                    + (size_t)(mTile << 4) * H4 + (nTile << 4) + l15;
#pragma unroll
    for (int r = 0; r < 8; ++r)
        out[(size_t)(r + half * 8) * H4] = acc[r];
}

// ---------------- WMMA LSTM scan: one block per direction, 2 chains in M rows 0..1 ----
// g[2,2048] = h[2,512] @ w_hh[d]^T computed on the matrix pipe each step.
__global__ __launch_bounds__(1024) void k_lstm_scan_wmma(const unsigned short* __restrict__ Whh,
                                                         const float* __restrict__ gx,
                                                         const float* __restrict__ b_ih,
                                                         const float* __restrict__ b_hh,
                                                         float* __restrict__ hout) {
    __shared__ alignas(16) unsigned short sh_h[2][HH];  // h per chain, bf16 (A fragments)
    __shared__ float sg[2][H4];                          // h @ W^T per chain
    __shared__ float sc[2][HH];                          // cell state per chain

    const int d    = blockIdx.x;                         // direction
    const int tid  = threadIdx.x;
    const int lane = tid & 31;
    const int wave = tid >> 5;                           // 0..31, owns 4 N-tiles
    const int l15  = lane & 15;
    const int half = lane >> 4;
    const int m    = l15;                                // A row this lane carries

    const unsigned short* W = Whh + (size_t)d * H4 * HH;

    // gate-stage identity: chain s = tid>>9, hidden unit jj = tid&511
    const int sch = tid >> 9;
    const int jj  = tid & (HH - 1);
    const int combo = sch * 2 + d;
    const float bi0 = b_ih[d * H4 + jj]          + b_hh[d * H4 + jj];
    const float bi1 = b_ih[d * H4 + HH + jj]     + b_hh[d * H4 + HH + jj];
    const float bi2 = b_ih[d * H4 + 2 * HH + jj] + b_hh[d * H4 + 2 * HH + jj];
    const float bi3 = b_ih[d * H4 + 3 * HH + jj] + b_hh[d * H4 + 3 * HH + jj];
    const float* gxc = gx + (size_t)combo * TT * H4;

    if (tid < 2 * HH) { sc[tid >> 9][tid & (HH - 1)] = 0.0f; sh_h[tid >> 9][tid & (HH - 1)] = 0; }
    __syncthreads();

    // B pointers for this wave's 4 N-tiles (B[k,n] = W[n,k] -> contiguous row reads)
    const unsigned short* brow0 = W + (size_t)((wave * 4 + 0) * 16 + l15) * HH + half * 16;
    const unsigned short* brow1 = W + (size_t)((wave * 4 + 1) * 16 + l15) * HH + half * 16;
    const unsigned short* brow2 = W + (size_t)((wave * 4 + 2) * 16 + l15) * HH + half * 16;
    const unsigned short* brow3 = W + (size_t)((wave * 4 + 3) * 16 + l15) * HH + half * 16;

    for (int t = 0; t < TT; ++t) {
        const int tg = d ? (TT - 1 - t) : t;

        v8f a0 = {}, a1 = {}, a2 = {}, a3 = {};
#pragma unroll 4
        for (int kb = 0; kb < HH; kb += 32) {
            // A fragment: rows 0..1 hold the two chains' h, rows 2..15 zero
            v16bf av = {};
            if (m < 2) {
                const unsigned short* hb = &sh_h[m][kb + half * 16];
                v8bf lo = *(const v8bf*)hb;
                v8bf hi = *(const v8bf*)(hb + 8);
                av = __builtin_shufflevector(lo, hi,
                        0,1,2,3,4,5,6,7,8,9,10,11,12,13,14,15);
            }
            v16bf b0 = *(const v16bf*)(brow0 + kb);
            v16bf b1 = *(const v16bf*)(brow1 + kb);
            v16bf b2 = *(const v16bf*)(brow2 + kb);
            v16bf b3 = *(const v16bf*)(brow3 + kb);
            a0 = __builtin_amdgcn_wmma_f32_16x16x32_bf16(false, av, false, b0, (short)0, a0, false, false);
            a1 = __builtin_amdgcn_wmma_f32_16x16x32_bf16(false, av, false, b1, (short)0, a1, false, false);
            a2 = __builtin_amdgcn_wmma_f32_16x16x32_bf16(false, av, false, b2, (short)0, a2, false, false);
            a3 = __builtin_amdgcn_wmma_f32_16x16x32_bf16(false, av, false, b3, (short)0, a3, false, false);
        }
        // D rows 0/1 live in accumulator elements 0/1 of lanes 0..15
        if (lane < 16) {
            sg[0][(wave * 4 + 0) * 16 + lane] = a0[0];
            sg[1][(wave * 4 + 0) * 16 + lane] = a0[1];
            sg[0][(wave * 4 + 1) * 16 + lane] = a1[0];
            sg[1][(wave * 4 + 1) * 16 + lane] = a1[1];
            sg[0][(wave * 4 + 2) * 16 + lane] = a2[0];
            sg[1][(wave * 4 + 2) * 16 + lane] = a2[1];
            sg[0][(wave * 4 + 3) * 16 + lane] = a3[0];
            sg[1][(wave * 4 + 3) * 16 + lane] = a3[1];
        }
        __syncthreads();

        // gates: 1024 threads = 2 chains x 512 hidden units
        {
            const float* gr = gxc + (size_t)tg * H4;
            float ii = gr[jj]          + bi0 + sg[sch][jj];
            float ff = gr[HH + jj]     + bi1 + sg[sch][HH + jj];
            float gg = gr[2 * HH + jj] + bi2 + sg[sch][2 * HH + jj];
            float oo = gr[3 * HH + jj] + bi3 + sg[sch][3 * HH + jj];
            float c = sigf(ff) * sc[sch][jj] + sigf(ii) * tanhf(gg);
            float h = sigf(oo) * tanhf(c);
            sc[sch][jj]   = c;
            sh_h[sch][jj] = f2bf(h);
            if (t == TT - 1) hout[combo * HH + jj] = h;
        }
        __syncthreads();
    }
}

// ---------------- epilogue: direction mix + bilinear + tanh + dot + sigmoid ----------------
__global__ __launch_bounds__(512) void k_final(const float* __restrict__ hout,
                                               const float* __restrict__ bi_w,
                                               const float* __restrict__ bi_b,
                                               const float* __restrict__ blA,
                                               const float* __restrict__ blB,
                                               const float* __restrict__ bl_b,
                                               const float* __restrict__ out_w,
                                               const float* __restrict__ out_b,
                                               float* __restrict__ out) {
    __shared__ float hA[HH];
    __shared__ float hB[HH];
    __shared__ float red[HH];
    const int j = threadIdx.x;
    const float w0 = bi_w[0], w1 = bi_w[1], bb = bi_b[0];
    // hout layout: combo = s*2 + d  ->  A: combos 0(fwd),1(bwd); B: combos 2(fwd),3(bwd)
    hA[j] = w0 * hout[0 * HH + j] + w1 * hout[1 * HH + j] + bb;
    hB[j] = w0 * hout[2 * HH + j] + w1 * hout[3 * HH + j] + bb;
    __syncthreads();
    float acc = bl_b[j];
    for (int k = 0; k < HH; ++k)
        acc += hA[k] * blA[(size_t)k * HH + j] + hB[k] * blB[(size_t)k * HH + j];
    red[j] = tanhf(acc) * out_w[j];
    __syncthreads();
    for (int s = HH / 2; s > 0; s >>= 1) {
        if (j < s) red[j] += red[j + s];
        __syncthreads();
    }
    if (j == 0) out[0] = sigf(red[0] + out_b[0]);
}

// ---------------- launcher ----------------
extern "C" void kernel_launch(void* const* d_in, const int* in_sizes, int n_in,
                              void* d_out, int out_size, void* d_ws, size_t ws_size,
                              hipStream_t stream) {
    const int*   sentA = (const int*)d_in[0];
    const int*   sentB = (const int*)d_in[1];
    const float* emb   = (const float*)d_in[3];
    const float* w_ih  = (const float*)d_in[4];
    const float* w_hh  = (const float*)d_in[5];
    const float* b_ih  = (const float*)d_in[6];
    const float* b_hh  = (const float*)d_in[7];
    const float* bi_w  = (const float*)d_in[8];
    const float* bi_b  = (const float*)d_in[9];
    const float* blA   = (const float*)d_in[10];
    const float* blB   = (const float*)d_in[11];
    const float* bl_b  = (const float*)d_in[12];
    const float* out_w = (const float*)d_in[13];
    const float* out_b = (const float*)d_in[14];
    float* out = (float*)d_out;

    char* ws = (char*)d_ws;
    unsigned short* Xbf  = (unsigned short*)(ws + OFF_X);
    unsigned short* Wih  = (unsigned short*)(ws + OFF_WIH);
    unsigned short* Whh  = (unsigned short*)(ws + OFF_WHH);
    float*          gx   = (float*)(ws + OFF_GX);
    float*          hout = (float*)(ws + OFF_HOUT);

    {
        int n = 2 * H4 * EP;
        k_conv_wih<<<(n + 255) / 256, 256, 0, stream>>>(w_ih, Wih);
    }
    {
        int n = 2 * H4 * HH;
        k_conv_whh<<<(n + 255) / 256, 256, 0, stream>>>(w_hh, Whh);
    }
    {
        int n = 2 * TT * EP;
        k_embed<<<(n + 255) / 256, 256, 0, stream>>>(sentA, sentB, emb, Xbf);
    }
    // WMMA GEMM: 4 combos x 128 x 128 tiles, 4 waves (tiles) per 128-thread block
    k_gemm_gx<<<16384, 128, 0, stream>>>(Xbf, Wih, gx);
    // 2 blocks (one per direction), each running both sentences' chains on the WMMA pipe
    k_lstm_scan_wmma<<<2, 1024, 0, stream>>>(Whh, gx, b_ih, b_hh, hout);
    // epilogue
    k_final<<<1, 512, 0, stream>>>(hout, bi_w, bi_b, blA, blB, bl_b, out_w, out_b, out);
}
